// MOE2_64364379898239
// MI455X (gfx1250) — matrix operand order repaired and checked by reading, not compiled, not run on hardware
//
#include <hip/hip_runtime.h>
#include <hip/hip_bf16.h>
#include <stdint.h>

#define N_NODES 40000
#define N_EDGES 500000
#define ET      (N_EDGES + N_NODES)   // 540000 edges incl. self loops
#define IN_DIM  128
#define ED_DIM  16
#define HID     64
#define HDIM    256                    // HEADS*HID = layer width
#define OUT_DIM 256
#define BATCH_B 128
#define NEG_SLOPE 0.2f

typedef __attribute__((ext_vector_type(16))) __bf16 v16bf;
typedef __attribute__((ext_vector_type(8)))  float  v8f;

#define CDIV(a, b) (((a) + (b) - 1) / (b))

// ---------------------------------------------------------------- utilities
__global__ void fill_f32(float* __restrict__ p, float v, long n) {
  long i = (long)blockIdx.x * blockDim.x + threadIdx.x;
  if (i < n) p[i] = v;
}
__global__ void fill_u32(unsigned* __restrict__ p, unsigned v, long n) {
  long i = (long)blockIdx.x * blockDim.x + threadIdx.x;
  if (i < n) p[i] = v;
}
__global__ void f2bf(const float* __restrict__ s, __bf16* __restrict__ d, long n) {
  long i = (long)blockIdx.x * blockDim.x + threadIdx.x;
  if (i < n) d[i] = (__bf16)s[i];
}

// -------------------------------------------------- self-loop construction
__global__ void edge_setup(const int* __restrict__ ei, const float* __restrict__ ea,
                           float* __restrict__ eaf, int* __restrict__ srcf,
                           int* __restrict__ dstf) {
  int e = blockIdx.x * blockDim.x + threadIdx.x;
  if (e >= N_EDGES) return;
  srcf[e] = ei[e];
  dstf[e] = ei[N_EDGES + e];
  const float4* s4 = (const float4*)(ea + (size_t)e * ED_DIM);
  float4* d4 = (float4*)(eaf + (size_t)e * ED_DIM);
  d4[0] = s4[0]; d4[1] = s4[1]; d4[2] = s4[2]; d4[3] = s4[3];
}

__global__ void loop_accum(const int* __restrict__ ei, const float* __restrict__ ea,
                           float* __restrict__ eaf, float* __restrict__ cnt) {
  int e = blockIdx.x * blockDim.x + threadIdx.x;
  if (e >= N_EDGES) return;
  int d = ei[N_EDGES + e];
  atomicAdd(&cnt[d], 1.0f);
  float* dst = eaf + (size_t)(N_EDGES + d) * ED_DIM;
  const float* src = ea + (size_t)e * ED_DIM;
#pragma unroll
  for (int j = 0; j < ED_DIM; ++j) atomicAdd(&dst[j], src[j]);
}

__global__ void loop_finalize(float* __restrict__ eaf, const float* __restrict__ cnt,
                              int* __restrict__ srcf, int* __restrict__ dstf) {
  int i = blockIdx.x * blockDim.x + threadIdx.x;
  if (i >= N_NODES) return;
  float inv = 1.0f / fmaxf(cnt[i], 1.0f);
  float* row = eaf + (size_t)(N_EDGES + i) * ED_DIM;
#pragma unroll
  for (int j = 0; j < ED_DIM; ++j) row[j] *= inv;
  srcf[N_EDGES + i] = i;
  dstf[N_EDGES + i] = i;
}

// -------------------------------------------------------- bf16 WMMA GEMM
// C[M,Nout] = A[M,K](bf16) @ B[K,Nout](bf16) + bias, optional relu.
// One wave -> 16x64 tile (4 accumulators reuse one A fragment).
// Software-pipelined: K-step k+1 fragments are fetched before the WMMAs of
// step k so VMEM overlaps the matrix pipe (partial loadcnt waits).
__device__ __forceinline__ v16bf load_a_frag(const __bf16* __restrict__ arow,
                                             int kk, int half) {
  const __bf16* ap = arow + kk + half * 8;
  v16bf a;
#pragma unroll
  for (int i = 0; i < 8; ++i) { a[i] = ap[i]; a[i + 8] = ap[16 + i]; }
  return a;
}

__device__ __forceinline__ v16bf load_b_frag(const __bf16* __restrict__ B, int kk, int half,
                                             int col, int Nout) {
  const __bf16* bp = B + (long)(kk + half * 8) * Nout + col;
  v16bf b;
#pragma unroll
  for (int i = 0; i < 8; ++i) { b[i] = bp[(long)i * Nout]; b[i + 8] = bp[(long)(16 + i) * Nout]; }
  return b;
}

__global__ void wmma_gemm_bias(const __bf16* __restrict__ A, const __bf16* __restrict__ B,
                               const float* __restrict__ bias, float* __restrict__ C,
                               int M, int K, int Nout, int relu) {
  const int lane = threadIdx.x & 31;
  const int wv   = threadIdx.x >> 5;
  const int mt   = blockIdx.x * 4 + wv;
  if (mt * 16 >= M) return;
  const int nb   = blockIdx.y * 64;
  const int half = lane >> 4;
  const int l16  = lane & 15;
  const long row = (long)(mt * 16 + l16);

  v8f acc0 = {}, acc1 = {}, acc2 = {}, acc3 = {};
  const __bf16* arow = A + row * K;

  // prologue: fragments for kk = 0
  v16bf a = load_a_frag(arow, 0, half);
  v16bf b0 = load_b_frag(B, 0, half, nb + l16, Nout);
  v16bf b1 = load_b_frag(B, 0, half, nb + l16 + 16, Nout);
  v16bf b2 = load_b_frag(B, 0, half, nb + l16 + 32, Nout);
  v16bf b3 = load_b_frag(B, 0, half, nb + l16 + 48, Nout);

  int kk = 0;
  for (; kk + 32 < K; kk += 32) {
    // prefetch next K-step before consuming current fragments
    v16bf an = load_a_frag(arow, kk + 32, half);
    v16bf nb0 = load_b_frag(B, kk + 32, half, nb + l16, Nout);
    v16bf nb1 = load_b_frag(B, kk + 32, half, nb + l16 + 16, Nout);
    v16bf nb2 = load_b_frag(B, kk + 32, half, nb + l16 + 32, Nout);
    v16bf nb3 = load_b_frag(B, kk + 32, half, nb + l16 + 48, Nout);

    acc0 = __builtin_amdgcn_wmma_f32_16x16x32_bf16(false, a, false, b0, (short)0, acc0, false, false);
    acc1 = __builtin_amdgcn_wmma_f32_16x16x32_bf16(false, a, false, b1, (short)0, acc1, false, false);
    acc2 = __builtin_amdgcn_wmma_f32_16x16x32_bf16(false, a, false, b2, (short)0, acc2, false, false);
    acc3 = __builtin_amdgcn_wmma_f32_16x16x32_bf16(false, a, false, b3, (short)0, acc3, false, false);

    a = an; b0 = nb0; b1 = nb1; b2 = nb2; b3 = nb3;
  }
  // epilogue: last K-step
  acc0 = __builtin_amdgcn_wmma_f32_16x16x32_bf16(false, a, false, b0, (short)0, acc0, false, false);
  acc1 = __builtin_amdgcn_wmma_f32_16x16x32_bf16(false, a, false, b1, (short)0, acc1, false, false);
  acc2 = __builtin_amdgcn_wmma_f32_16x16x32_bf16(false, a, false, b2, (short)0, acc2, false, false);
  acc3 = __builtin_amdgcn_wmma_f32_16x16x32_bf16(false, a, false, b3, (short)0, acc3, false, false);

  // C/D layout: VGPR r -> m = r + 8*(lane/16), n = lane%16
#pragma unroll
  for (int r = 0; r < 8; ++r) {
    const long m = (long)(mt * 16 + r + 8 * half);
    float* cr = C + m * Nout;
    const int n0 = nb + l16;
    float v0 = acc0[r] + bias[n0];
    float v1 = acc1[r] + bias[n0 + 16];
    float v2 = acc2[r] + bias[n0 + 32];
    float v3 = acc3[r] + bias[n0 + 48];
    if (relu) {
      v0 = fmaxf(v0, 0.f); v1 = fmaxf(v1, 0.f);
      v2 = fmaxf(v2, 0.f); v3 = fmaxf(v3, 0.f);
    }
    cr[n0] = v0; cr[n0 + 16] = v1; cr[n0 + 32] = v2; cr[n0 + 48] = v3;
  }
}

// ------------------------------------------------------------ edge kernels
// logit[e,h] = sum_c leaky_relu(xl[s,h,c]+xr[d,h,c]+ (ea[e,:]@We)[h,c]) * att[h,c]
template <int HEADS, int C>
__global__ void edge_logits_k(const int* __restrict__ srcf, const int* __restrict__ dstf,
                              const float* __restrict__ eaf,
                              const float* __restrict__ xl, const float* __restrict__ xr,
                              const float* __restrict__ We, const float* __restrict__ att,
                              float* __restrict__ logits, int npairs) {
  constexpr int H = HEADS * C;
  const int lane = threadIdx.x & 31;
  int p = (int)(((long)blockIdx.x * blockDim.x + threadIdx.x) >> 5);
  if (p >= npairs) return;
  const int e = p / HEADS;
  const int h = p - e * HEADS;
  const int s = srcf[e];
  const int d = dstf[e];
  float ear[ED_DIM];
  const float4* e4 = (const float4*)(eaf + (size_t)e * ED_DIM);
#pragma unroll
  for (int q = 0; q < 4; ++q) {
    float4 t = e4[q];
    ear[4 * q] = t.x; ear[4 * q + 1] = t.y; ear[4 * q + 2] = t.z; ear[4 * q + 3] = t.w;
  }
  const float* xls = xl + (size_t)s * H + h * C;
  const float* xrd = xr + (size_t)d * H + h * C;
  const float* weh = We + h * C;   // We is [ED_DIM, H]
  const float* ath = att + h * C;
  float acc = 0.f;
#pragma unroll
  for (int k = 0; k < C / 32; ++k) {
    const int c = lane + 32 * k;
    float ef = 0.f;
#pragma unroll
    for (int j = 0; j < ED_DIM; ++j) ef = fmaf(ear[j], weh[(size_t)j * H + c], ef);
    float t = xls[c] + xrd[c] + ef;
    t = t > 0.f ? t : NEG_SLOPE * t;
    acc = fmaf(t, ath[c], acc);
  }
#pragma unroll
  for (int off = 16; off > 0; off >>= 1) acc += __shfl_xor(acc, off, 32);
  if (lane == 0) logits[p] = acc;
}

// segment max via monotone-uint atomicMax (self-loops => no empty segments)
__global__ void seg_max_k(const int* __restrict__ seg, const float* __restrict__ vals,
                          unsigned* __restrict__ mu, int npairs, int heads) {
  int p = blockIdx.x * blockDim.x + threadIdx.x;
  if (p >= npairs) return;
  int e = p / heads, h = p - e * heads;
  int d = seg[e];
  unsigned b = __float_as_uint(vals[p]);
  unsigned u = (b & 0x80000000u) ? ~b : (b | 0x80000000u);
  atomicMax(&mu[(size_t)d * heads + h], u);
}

__global__ void seg_exp_k(const int* __restrict__ seg, const float* __restrict__ vals,
                          const unsigned* __restrict__ mu, float* __restrict__ evals,
                          float* __restrict__ zz, int npairs, int heads) {
  int p = blockIdx.x * blockDim.x + threadIdx.x;
  if (p >= npairs) return;
  int e = p / heads, h = p - e * heads;
  int d = seg[e];
  unsigned u = mu[(size_t)d * heads + h];
  unsigned b = (u & 0x80000000u) ? (u & 0x7FFFFFFFu) : ~u;
  float m = __uint_as_float(b);
  float f = __expf(vals[p] - m);
  evals[p] = f;
  atomicAdd(&zz[(size_t)d * heads + h], f);
}

template <int HEADS, int C>
__global__ void edge_scatter_k(const int* __restrict__ srcf, const int* __restrict__ dstf,
                               const float* __restrict__ xl, const float* __restrict__ evals,
                               const float* __restrict__ zz, float* __restrict__ out,
                               int npairs) {
  constexpr int H = HEADS * C;
  const int lane = threadIdx.x & 31;
  int p = (int)(((long)blockIdx.x * blockDim.x + threadIdx.x) >> 5);
  if (p >= npairs) return;
  const int e = p / HEADS;
  const int h = p - e * HEADS;
  const int s = srcf[e];
  const int d = dstf[e];
  const float alpha = evals[p] / (zz[(size_t)d * HEADS + h] + 1e-16f);
  const float* xls = xl + (size_t)s * H + h * C;
  float* od = out + (size_t)d * H + h * C;
#pragma unroll
  for (int k = 0; k < C / 32; ++k) {
    const int c = lane + 32 * k;
    atomicAdd(&od[c], xls[c] * alpha);
  }
}

__global__ void bias_act_k(float* __restrict__ out, const float* __restrict__ bias,
                           long total, int H, int relu) {
  long t = (long)blockIdx.x * blockDim.x + threadIdx.x;
  if (t >= total) return;
  int col = (int)(t % H);
  float v = out[t] + bias[col];
  if (relu) v = fmaxf(v, 0.f);
  out[t] = v;
}

// ------------------------------------------------------------ pooling/head
__global__ void gate_dot_k(const float* __restrict__ gtmp, const float* __restrict__ W2,
                           const float* __restrict__ b2, float* __restrict__ gate) {
  const int lane = threadIdx.x & 31;
  int i = (int)(((long)blockIdx.x * blockDim.x + threadIdx.x) >> 5);
  if (i >= N_NODES) return;
  const float4 g4 = *(const float4*)(gtmp + (size_t)i * 128 + lane * 4);
  const float4 w4 = *(const float4*)(W2 + lane * 4);
  float acc = g4.x * w4.x + g4.y * w4.y + g4.z * w4.z + g4.w * w4.w;
#pragma unroll
  for (int off = 16; off > 0; off >>= 1) acc += __shfl_xor(acc, off, 32);
  if (lane == 0) gate[i] = acc + b2[0];
}

__global__ void pool_scatter_k(const int* __restrict__ batch, const float* __restrict__ h3,
                               const float* __restrict__ ae, const float* __restrict__ pz,
                               float* __restrict__ gacc) {
  const int lane = threadIdx.x & 31;
  int i = (int)(((long)blockIdx.x * blockDim.x + threadIdx.x) >> 5);
  if (i >= N_NODES) return;
  const int b = batch[i];
  const float coef = ae[i] / (pz[b] + 1e-16f);
#pragma unroll
  for (int k = 0; k < OUT_DIM / 32; ++k) {
    const int c = lane + 32 * k;
    atomicAdd(&gacc[(size_t)b * OUT_DIM + c], h3[(size_t)i * OUT_DIM + c] * coef);
  }
}

__global__ void final_mlp_k(const float* __restrict__ g, const float* __restrict__ W1,
                            const float* __restrict__ b1, const float* __restrict__ W2,
                            const float* __restrict__ b2, float* __restrict__ out) {
  __shared__ float t[128];
  const int b = blockIdx.x;
  const int j = threadIdx.x;
  float acc = b1[j];
  const float* gr = g + (size_t)b * OUT_DIM;
  for (int i = 0; i < OUT_DIM; ++i) acc = fmaf(gr[i], W1[(size_t)i * 128 + j], acc);
  t[j] = fmaxf(acc, 0.f);
  __syncthreads();
  if (j < 2) {
    float o = b2[j];
    for (int i = 0; i < 128; ++i) o = fmaf(t[i], W2[i * 2 + j], o);
    out[b * 2 + j] = o;
  }
}

// ------------------------------------------------------------------- host
extern "C" void kernel_launch(void* const* d_in, const int* in_sizes, int n_in,
                              void* d_out, int out_size, void* d_ws, size_t ws_size,
                              hipStream_t stream) {
  (void)in_sizes; (void)n_in; (void)out_size; (void)ws_size;
  const float* x      = (const float*)d_in[0];
  const int*   ei     = (const int*)d_in[1];
  const float* ea     = (const float*)d_in[2];
  const int*   batch  = (const int*)d_in[3];
  const float* c_Wl[3]   = {(const float*)d_in[5],  (const float*)d_in[12], (const float*)d_in[19]};
  const float* c_bl[3]   = {(const float*)d_in[6],  (const float*)d_in[13], (const float*)d_in[20]};
  const float* c_Wr[3]   = {(const float*)d_in[7],  (const float*)d_in[14], (const float*)d_in[21]};
  const float* c_br[3]   = {(const float*)d_in[8],  (const float*)d_in[15], (const float*)d_in[22]};
  const float* c_We[3]   = {(const float*)d_in[9],  (const float*)d_in[16], (const float*)d_in[23]};
  const float* c_att[3]  = {(const float*)d_in[10], (const float*)d_in[17], (const float*)d_in[24]};
  const float* c_bias[3] = {(const float*)d_in[11], (const float*)d_in[18], (const float*)d_in[25]};
  const float* gate_W1 = (const float*)d_in[26];
  const float* gate_b1 = (const float*)d_in[27];
  const float* gate_W2 = (const float*)d_in[28];
  const float* gate_b2 = (const float*)d_in[29];
  const float* reg_W1  = (const float*)d_in[30];
  const float* reg_b1  = (const float*)d_in[31];
  const float* reg_W2  = (const float*)d_in[32];
  const float* reg_b2  = (const float*)d_in[33];
  float* outp = (float*)d_out;

  // ---- workspace layout (256B aligned regions)
  char* base = (char*)d_ws;
  size_t off = 0;
  auto alloc = [&](size_t bytes) -> char* {
    char* p = base + off;
    off += (bytes + 255) & ~(size_t)255;
    return p;
  };
  float*    eaf   = (float*)alloc((size_t)ET * ED_DIM * 4);
  int*      srcf  = (int*)alloc((size_t)ET * 4);
  int*      dstf  = (int*)alloc((size_t)ET * 4);
  float*    cnt   = (float*)alloc((size_t)N_NODES * 4);
  __bf16*   featb = (__bf16*)alloc((size_t)N_NODES * HDIM * 2);
  __bf16*   wbA   = (__bf16*)alloc((size_t)HDIM * HDIM * 2);
  __bf16*   wbB   = (__bf16*)alloc((size_t)HDIM * HDIM * 2);
  float*    xl    = (float*)alloc((size_t)N_NODES * HDIM * 4);
  float*    xr    = (float*)alloc((size_t)N_NODES * HDIM * 4);
  float*    featA = (float*)alloc((size_t)N_NODES * HDIM * 4);
  float*    featB = (float*)alloc((size_t)N_NODES * HDIM * 4);
  float*    logits= (float*)alloc((size_t)ET * 4 * 4);
  float*    evals = (float*)alloc((size_t)ET * 4 * 4);
  unsigned* mu    = (unsigned*)alloc((size_t)N_NODES * 4 * 4);
  float*    zz    = (float*)alloc((size_t)N_NODES * 4 * 4);
  float*    gate  = (float*)alloc((size_t)N_NODES * 4);
  float*    ae    = (float*)alloc((size_t)N_NODES * 4);
  unsigned* pmu   = (unsigned*)alloc((size_t)BATCH_B * 4);
  float*    pz    = (float*)alloc((size_t)BATCH_B * 4);
  float*    gacc  = (float*)alloc((size_t)BATCH_B * OUT_DIM * 4);
  float*    gtmp  = xl;  // reuse: xl free during pooling (N x 128 <= N x 256)

  const int TB = 256;

  // ---- self-loop mean edge attrs + full edge lists
  fill_f32<<<CDIV((long)N_NODES * ED_DIM, TB), TB, 0, stream>>>(eaf + (size_t)N_EDGES * ED_DIM, 0.f, (long)N_NODES * ED_DIM);
  fill_f32<<<CDIV(N_NODES, TB), TB, 0, stream>>>(cnt, 0.f, N_NODES);
  edge_setup<<<CDIV(N_EDGES, TB), TB, 0, stream>>>(ei, ea, eaf, srcf, dstf);
  loop_accum<<<CDIV(N_EDGES, TB), TB, 0, stream>>>(ei, ea, eaf, cnt);
  loop_finalize<<<CDIV(N_NODES, TB), TB, 0, stream>>>(eaf, cnt, srcf, dstf);

  // ---- GATv2 layers
  auto run_layer = [&](const float* fin, int K, int li, int heads, float* fout, int relu) {
    const int H = HDIM;                         // all layers output 256
    const long nfeat = (long)N_NODES * K;
    f2bf<<<CDIV(nfeat, TB), TB, 0, stream>>>(fin, featb, nfeat);
    f2bf<<<CDIV((long)K * H, TB), TB, 0, stream>>>(c_Wl[li], wbA, (long)K * H);
    f2bf<<<CDIV((long)K * H, TB), TB, 0, stream>>>(c_Wr[li], wbB, (long)K * H);
    dim3 gg(CDIV(N_NODES / 16, 4), H / 64);
    wmma_gemm_bias<<<gg, 128, 0, stream>>>(featb, wbA, c_bl[li], xl, N_NODES, K, H, 0);
    wmma_gemm_bias<<<gg, 128, 0, stream>>>(featb, wbB, c_br[li], xr, N_NODES, K, H, 0);

    fill_f32<<<CDIV((long)N_NODES * H, TB), TB, 0, stream>>>(fout, 0.f, (long)N_NODES * H);
    fill_u32<<<CDIV((long)N_NODES * heads, TB), TB, 0, stream>>>(mu, 0u, (long)N_NODES * heads);
    fill_f32<<<CDIV((long)N_NODES * heads, TB), TB, 0, stream>>>(zz, 0.f, (long)N_NODES * heads);

    const int npairs = ET * heads;
    const int wblk = CDIV(npairs * 32, TB);
    if (heads == 4) {
      edge_logits_k<4, 64><<<wblk, TB, 0, stream>>>(srcf, dstf, eaf, xl, xr, c_We[li], c_att[li], logits, npairs);
    } else {
      edge_logits_k<1, 256><<<wblk, TB, 0, stream>>>(srcf, dstf, eaf, xl, xr, c_We[li], c_att[li], logits, npairs);
    }
    seg_max_k<<<CDIV(npairs, TB), TB, 0, stream>>>(dstf, logits, mu, npairs, heads);
    seg_exp_k<<<CDIV(npairs, TB), TB, 0, stream>>>(dstf, logits, mu, evals, zz, npairs, heads);
    if (heads == 4) {
      edge_scatter_k<4, 64><<<wblk, TB, 0, stream>>>(srcf, dstf, xl, evals, zz, fout, npairs);
    } else {
      edge_scatter_k<1, 256><<<wblk, TB, 0, stream>>>(srcf, dstf, xl, evals, zz, fout, npairs);
    }
    bias_act_k<<<CDIV((long)N_NODES * H, TB), TB, 0, stream>>>(fout, c_bias[li], (long)N_NODES * H, H, relu);
  };

  run_layer(x,     IN_DIM, 0, 4, featA, 1);   // conv1 + relu
  run_layer(featA, HDIM,   1, 4, featB, 1);   // conv2 + relu
  run_layer(featB, HDIM,   2, 1, featA, 0);   // conv3 (heads=1, no relu) -> h3 in featA

  // ---- attentional aggregation pooling
  const float* h3 = featA;
  f2bf<<<CDIV((long)N_NODES * OUT_DIM, TB), TB, 0, stream>>>(h3, featb, (long)N_NODES * OUT_DIM);
  f2bf<<<CDIV((long)OUT_DIM * 128, TB), TB, 0, stream>>>(gate_W1, wbA, (long)OUT_DIM * 128);
  dim3 gg(CDIV(N_NODES / 16, 4), 128 / 64);
  wmma_gemm_bias<<<gg, 128, 0, stream>>>(featb, wbA, gate_b1, gtmp, N_NODES, OUT_DIM, 128, 1);
  gate_dot_k<<<CDIV(N_NODES * 32, TB), TB, 0, stream>>>(gtmp, gate_W2, gate_b2, gate);

  fill_u32<<<1, BATCH_B, 0, stream>>>(pmu, 0u, BATCH_B);
  fill_f32<<<1, BATCH_B, 0, stream>>>(pz, 0.f, BATCH_B);
  fill_f32<<<CDIV((long)BATCH_B * OUT_DIM, TB), TB, 0, stream>>>(gacc, 0.f, (long)BATCH_B * OUT_DIM);
  seg_max_k<<<CDIV(N_NODES, TB), TB, 0, stream>>>(batch, gate, pmu, N_NODES, 1);
  seg_exp_k<<<CDIV(N_NODES, TB), TB, 0, stream>>>(batch, gate, pmu, ae, pz, N_NODES, 1);
  pool_scatter_k<<<CDIV(N_NODES * 32, TB), TB, 0, stream>>>(batch, h3, ae, pz, gacc);

  // ---- regression head
  final_mlp_k<<<BATCH_B, 128, 0, stream>>>(gacc, reg_W1, reg_b1, reg_W2, reg_b2, outp);
}
